// COACNNet_77146202571253
// MI455X (gfx1250) — matrix operand order repaired and checked by reading, not compiled, not run on hardware
//
#include <hip/hip_runtime.h>
#include <math.h>

// ---------------------------------------------------------------------------
// COACNNet pipeline for MI455X (gfx1250, wave32, WMMA)
//  - GEMMs: V_WMMA_F32_16X16X32_F16; gemm waves compute 32x64 tiles (MT=2 x
//    NT=4 accumulators, 8 independent WMMAs/k-step), b128 vector loads,
//    pre-transposed weights for coalesced B^T access. All tile counts are
//    compile-time: no divergent control flow around WMMA.
//  - LightGCN: bandwidth-bound gather + global_atomic_add_f32 scatter.
// ---------------------------------------------------------------------------

typedef __attribute__((ext_vector_type(16))) _Float16 v16h;
typedef __attribute__((ext_vector_type(8)))  float    v8f;

#define NUM_MASHUP 50000
#define NUM_API    50000
#define NUM_DOMAIN 500
#define FEAT       128
#define MC         768
#define NN         (NUM_MASHUP + NUM_API)

__device__ __forceinline__ float sigmoidf_(float v) {
    return __builtin_amdgcn_rcpf(1.0f + __expf(-v));   // v_exp_f32 + v_rcp_f32
}

// ---------------------------------------------------------------------------
// WMMA lane layouts per CDNA5 ISA 7.12.2 (V_WMMA_F32_16X16X32_F16):
//   A (16x32 f16):  m = lane%16 ; half h -> k = (h&7) + ((h&8)<<1) + (hi ? 8 : 0)
//                   => code uses contiguous k = h + (hi ? 16 : 0) instead: one 16-float run
//   B (32x16 f16):  n = lane%16 ; same table k order as A; code loads the same contiguous k
//                   => one contiguous 16-float run (B^T row-major)
//   C/D (16x16 f32): n = lane%16 ; vgpr r -> m = r + (hi ? 8 : 0)
// ---------------------------------------------------------------------------

__device__ __forceinline__ v16h load_a_frag(const float* __restrict__ A, size_t rowoff,
                                            int k0, bool hi) {
    const float4* pa = (const float4*)(A + rowoff + k0 + (hi ? 16 : 0));
    float4 a0 = pa[0], a1 = pa[1];   // k offsets 0..7
    float4 a2 = pa[2], a3 = pa[3];   // k offsets 8..15
    v16h a;
    a[0]  = (_Float16)a0.x; a[1]  = (_Float16)a0.y; a[2]  = (_Float16)a0.z; a[3]  = (_Float16)a0.w;
    a[4]  = (_Float16)a1.x; a[5]  = (_Float16)a1.y; a[6]  = (_Float16)a1.z; a[7]  = (_Float16)a1.w;
    a[8]  = (_Float16)a2.x; a[9]  = (_Float16)a2.y; a[10] = (_Float16)a2.z; a[11] = (_Float16)a2.w;
    a[12] = (_Float16)a3.x; a[13] = (_Float16)a3.y; a[14] = (_Float16)a3.z; a[15] = (_Float16)a3.w;
    return a;
}

__device__ __forceinline__ v16h load_bt_frag(const float* __restrict__ B, size_t rowoff,
                                             int k0, bool hi) {
    const float4* pb = (const float4*)(B + rowoff + k0 + (hi ? 16 : 0));
    float4 b0 = pb[0], b1 = pb[1], b2 = pb[2], b3 = pb[3];  // 16 contiguous floats
    v16h b;
    b[0]  = (_Float16)b0.x; b[1]  = (_Float16)b0.y; b[2]  = (_Float16)b0.z; b[3]  = (_Float16)b0.w;
    b[4]  = (_Float16)b1.x; b[5]  = (_Float16)b1.y; b[6]  = (_Float16)b1.z; b[7]  = (_Float16)b1.w;
    b[8]  = (_Float16)b2.x; b[9]  = (_Float16)b2.y; b[10] = (_Float16)b2.z; b[11] = (_Float16)b2.w;
    b[12] = (_Float16)b3.x; b[13] = (_Float16)b3.y; b[14] = (_Float16)b3.z; b[15] = (_Float16)b3.w;
    return b;
}

// ---------------------------------------------------------------------------
// Y = sigmoid(X @ Wt^T + bias); Wt is [FEAT, K] pre-transposed; MT=1, NT=4.
// blockDim = (32,2): 2 waves cover all FEAT=128 columns; no tile guards.
// Row loads clamped to rowLimit-1 (padded rows get finite dummy values).
// ---------------------------------------------------------------------------
__global__ void siglin_wmma_kernel(const float* __restrict__ X, int lda,
                                   const float* __restrict__ Wt,    // [FEAT, K]
                                   const float* __restrict__ bias,  // [FEAT]
                                   float* __restrict__ Y,
                                   int K, int rowLimit) {
    const int  lane = threadIdx.x & 31;
    const int  l16  = lane & 15;
    const bool hi   = lane >= 16;
    const int  mbase = blockIdx.x * 16;
    const int  nbase = threadIdx.y * 64;             // 0 or 64

    int arow = mbase + l16;
    if (arow > rowLimit - 1) arow = rowLimit - 1;
    const size_t aoff = (size_t)arow * lda;

    v8f acc[4] = {};
    for (int k0 = 0; k0 < K; k0 += 32) {
        v16h a = load_a_frag(X, aoff, k0, hi);
#pragma unroll
        for (int t = 0; t < 4; ++t) {
            v16h b = load_bt_frag(Wt, (size_t)(nbase + t * 16 + l16) * K, k0, hi);
            acc[t] = __builtin_amdgcn_wmma_f32_16x16x32_f16(
                false, a, false, b, (short)0, acc[t], false, false);
        }
    }

#pragma unroll
    for (int t = 0; t < 4; ++t) {
        const int   n  = nbase + t * 16 + l16;
        const float bn = bias[n];
#pragma unroll
        for (int r = 0; r < 8; ++r) {
            int m = mbase + r + (hi ? 8 : 0);
            Y[(size_t)m * FEAT + n] = sigmoidf_(acc[t][r] + bn);
        }
    }
}

// ---------------------------------------------------------------------------
// C = scale*(A @ B[^T]) [+ dscale*D].  MT=2 (32 rows), NTILES x 16 cols.
// nbase = ncol0 + group*16*NTILES; wave exits (uniformly) if nbase >= nlimit.
// All processed groups are full: no guards inside the k-loop.
// ---------------------------------------------------------------------------
template<bool BT, int NTILES>
__global__ void gemm_wmma_kernel(const float* __restrict__ A, int lda,
                                 const float* __restrict__ B, int ldb,
                                 float* __restrict__ C, int ldc,
                                 int K, int ncol0, int nlimit, float scale,
                                 const float* __restrict__ D, int ldd, float dscale) {
    const int  lane = threadIdx.x & 31;
    const int  l16  = lane & 15;
    const bool hi   = lane >= 16;
    const int  mbase = blockIdx.x * 32;
    const int  group = blockIdx.y * blockDim.y + threadIdx.y;
    const int  nbase = ncol0 + group * (16 * NTILES);
    if (nbase >= nlimit) return;                     // wave-uniform whole-body skip

    const size_t aoff0 = (size_t)(mbase + l16) * lda;
    const size_t aoff1 = (size_t)(mbase + 16 + l16) * lda;

    v8f acc[2][NTILES];
#pragma unroll
    for (int mt = 0; mt < 2; ++mt)
#pragma unroll
        for (int t = 0; t < NTILES; ++t) acc[mt][t] = (v8f){};

    for (int k0 = 0; k0 < K; k0 += 32) {
        v16h a0 = load_a_frag(A, aoff0, k0, hi);
        v16h a1 = load_a_frag(A, aoff1, k0, hi);
#pragma unroll
        for (int t = 0; t < NTILES; ++t) {
            v16h b;
            if (BT) {
                b = load_bt_frag(B, (size_t)(nbase + t * 16 + l16) * ldb, k0, hi);
            } else {
#pragma unroll
                for (int h = 0; h < 16; ++h) {
                    int kb = k0 + h + (hi ? 16 : 0);
                    b[h] = (_Float16)B[(size_t)kb * ldb + (nbase + t * 16 + l16)];
                }
            }
            acc[0][t] = __builtin_amdgcn_wmma_f32_16x16x32_f16(
                false, a0, false, b, (short)0, acc[0][t], false, false);
            acc[1][t] = __builtin_amdgcn_wmma_f32_16x16x32_f16(
                false, a1, false, b, (short)0, acc[1][t], false, false);
        }
    }

#pragma unroll
    for (int mt = 0; mt < 2; ++mt) {
#pragma unroll
        for (int t = 0; t < NTILES; ++t) {
            const int n = nbase + t * 16 + l16;
#pragma unroll
            for (int r = 0; r < 8; ++r) {
                int   m = mbase + mt * 16 + r + (hi ? 8 : 0);
                float v = scale * acc[mt][t][r];
                if (D) v += dscale * D[(size_t)m * ldd + n];
                C[(size_t)m * ldc + n] = v;
            }
        }
    }
}

// Wt[n*K + k] = W[k*FEAT + n]
__global__ void transpose_w_kernel(const float* __restrict__ W, float* __restrict__ Wt, int K) {
    int i = blockIdx.x * blockDim.x + threadIdx.x;       // i = n*K + k
    if (i >= K * FEAT) return;
    int n = i / K, k = i - n * K;
    Wt[i] = W[(size_t)k * FEAT + n];
}

// alpha[r,c] = al[r,c] / sum_{c<cols} al[r,c]  (cols..ldp-1 zeroed), in place.
__global__ void rownorm_kernel(float* __restrict__ al, int cols, int ldp, int rows) {
    int r = blockIdx.x * blockDim.x + threadIdx.x;
    if (r >= rows) return;
    float s = 0.f;
    for (int c = 0; c < cols; ++c) s += al[(size_t)r * ldp + c];
    float inv = 1.0f / s;
    for (int c = 0; c < ldp; ++c) {
        float v = (c < cols) ? al[(size_t)r * ldp + c] * inv : 0.f;
        al[(size_t)r * ldp + c] = v;
    }
}

__global__ void fill_kernel(float* __restrict__ p, float v, size_t n) {
    size_t i = blockIdx.x * (size_t)blockDim.x + threadIdx.x;
    if (i < n) p[i] = v;
}

__global__ void scale_copy_kernel(float* __restrict__ y, const float* __restrict__ x,
                                  float a, size_t n) {
    size_t i = blockIdx.x * (size_t)blockDim.x + threadIdx.x;
    if (i < n) y[i] = a * x[i];
}

__global__ void axpy_kernel(float* __restrict__ y, const float* __restrict__ x,
                            float a, size_t n) {
    size_t i = blockIdx.x * (size_t)blockDim.x + threadIdx.x;
    if (i < n) y[i] += a * x[i];
}

// degree count over row = concat(src, NM + dst)
__global__ void deg_kernel(const int* __restrict__ src, const int* __restrict__ dst,
                           float* __restrict__ deg, int E) {
    size_t i = blockIdx.x * (size_t)blockDim.x + threadIdx.x;
    if (i >= 2 * (size_t)E) return;
    int node = (i < (size_t)E) ? src[i] : (NUM_MASHUP + dst[i - (size_t)E]);
    atomicAdd(&deg[node], 1.0f);
}

__global__ void dinv_kernel(float* __restrict__ deg, int n) {
    int i = blockIdx.x * blockDim.x + threadIdx.x;
    if (i >= n) return;
    float d = deg[i];
    deg[i] = (d > 0.f) ? rsqrtf(d) : 0.f;
}

// one wave per directed edge; lane covers 4 consecutive features (float4 gather)
__global__ void scatter_kernel(const int* __restrict__ src, const int* __restrict__ dst,
                               const float* __restrict__ dinv,
                               const float* __restrict__ xin, float* __restrict__ xout,
                               int E) {
    size_t t    = blockIdx.x * (size_t)blockDim.x + threadIdx.x;
    size_t edge = t >> 5;
    int    lane = (int)(t & 31);
    if (edge >= 2 * (size_t)E) return;
    int row, col;
    if (edge < (size_t)E) { row = src[edge];             col = NUM_MASHUP + dst[edge]; }
    else                  { size_t e = edge - (size_t)E; row = NUM_MASHUP + dst[e]; col = src[e]; }
    float nrm = dinv[row] * dinv[col];
    const float4 v = ((const float4*)(xin + (size_t)row * FEAT))[lane];
    float* o = xout + (size_t)col * FEAT + 4 * lane;
    atomicAdd(o + 0, v.x * nrm);
    atomicAdd(o + 1, v.y * nrm);
    atomicAdd(o + 2, v.z * nrm);
    atomicAdd(o + 3, v.w * nrm);
}

// ---------------------------------------------------------------------------

static inline dim3 g1(size_t n, int b) { return dim3((unsigned)((n + b - 1) / b)); }

extern "C" void kernel_launch(void* const* d_in, const int* in_sizes, int n_in,
                              void* d_out, int out_size, void* d_ws, size_t ws_size,
                              hipStream_t stream) {
    (void)n_in; (void)out_size; (void)ws_size;

    const float* x    = (const float*)d_in[0];
    const float* me   = (const float*)d_in[1];
    const float* de   = (const float*)d_in[2];
    const float* ae   = (const float*)d_in[3];
    const float* Wsde = (const float*)d_in[4];
    const float* bsde = (const float*)d_in[5];
    const float* Wval = (const float*)d_in[6];
    const float* bval = (const float*)d_in[7];
    const float* Wkey = (const float*)d_in[8];
    const float* bkey = (const float*)d_in[9];
    const float* Wsie = (const float*)d_in[10];
    const float* bsie = (const float*)d_in[11];
    const int*   esrc = (const int*)d_in[12];
    const int*   edst = (const int*)d_in[13];
    const int    E     = in_sizes[12];
    const int    BATCH = in_sizes[0] / MC;          // 1024
    const int    NDP   = 512;                       // NUM_DOMAIN padded to x16

    // workspace layout (floats; all offsets multiples of 4 -> 16B aligned)
    float* ws   = (float*)d_ws;
    size_t off  = 0;
    float* xa    = ws + off; off += (size_t)NN * FEAT;     // emb / layer ping
    float* xb    = ws + off; off += (size_t)NN * FEAT;     // layer pong
    float* outA  = ws + off; off += (size_t)NN * FEAT;     // lightgcn accumulator
    float* deg   = ws + off; off += (size_t)NN;            // degree -> dinv
    float* vmi   = ws + off; off += (size_t)BATCH * FEAT;
    float* vkey  = ws + off; off += (size_t)NDP * FEAT;    // padded rows
    float* vval  = ws + off; off += (size_t)NDP * FEAT;
    float* alp   = ws + off; off += (size_t)BATCH * NDP;   // al -> alpha in place
    float* zm    = ws + off; off += (size_t)BATCH * FEAT;
    float* WsdeT = ws + off; off += (size_t)FEAT * MC;     // transposed weights [128,768]
    float* WsieT = ws + off; off += (size_t)FEAT * MC;
    float* WkeyT = ws + off; off += (size_t)FEAT * MC;
    float* WvalT = ws + off; off += (size_t)FEAT * MC;
    float* pred  = (float*)d_out;

    // 0) transpose weights to [FEAT, K] for coalesced B^T loads ------------------
    transpose_w_kernel<<<g1((size_t)FEAT * MC, 256), dim3(256), 0, stream>>>(Wsde, WsdeT, MC);
    transpose_w_kernel<<<g1((size_t)FEAT * MC, 256), dim3(256), 0, stream>>>(Wsie, WsieT, MC);
    transpose_w_kernel<<<g1((size_t)FEAT * MC, 256), dim3(256), 0, stream>>>(Wkey, WkeyT, MC);
    transpose_w_kernel<<<g1((size_t)FEAT * MC, 256), dim3(256), 0, stream>>>(Wval, WvalT, MC);

    // 1) sigmoid-linear features (blockDim (32,2): 2 waves cover 128 cols) -------
    const dim3 sblk(32, 2, 1);
    siglin_wmma_kernel<<<dim3(BATCH / 16, 1), sblk, 0, stream>>>(
        x, MC, WsdeT, bsde, vmi, MC, BATCH);
    siglin_wmma_kernel<<<dim3(NDP / 16, 1), sblk, 0, stream>>>(
        de, MC, WkeyT, bkey, vkey, MC, NUM_DOMAIN);
    siglin_wmma_kernel<<<dim3(NDP / 16, 1), sblk, 0, stream>>>(
        de, MC, WvalT, bval, vval, MC, NUM_DOMAIN);
    siglin_wmma_kernel<<<dim3(NUM_MASHUP / 16, 1), sblk, 0, stream>>>(
        me, MC, WsdeT, bsde, xa, MC, NUM_MASHUP);
    siglin_wmma_kernel<<<dim3(NUM_API / 16, 1), sblk, 0, stream>>>(
        ae, MC, WsieT, bsie, xa + (size_t)NUM_MASHUP * FEAT, MC, NUM_API);

    // 2) attention: al = v_mi @ v_key^T ; rownorm ; z_m = 0.5*(alpha@v_val)+0.5*v_mi
    gemm_wmma_kernel<true, 4><<<dim3(BATCH / 32, 2), dim3(32, 4), 0, stream>>>(
        vmi, FEAT, vkey, FEAT, alp, NDP, FEAT, 0, NDP, 1.f, nullptr, 0, 0.f);
    rownorm_kernel<<<g1(BATCH, 256), dim3(256), 0, stream>>>(alp, NUM_DOMAIN, NDP, BATCH);
    gemm_wmma_kernel<false, 4><<<dim3(BATCH / 32, 1), dim3(32, 2), 0, stream>>>(
        alp, NDP, vval, FEAT, zm, FEAT, NDP, 0, FEAT, 0.5f, vmi, FEAT, 0.5f);

    // 3) LightGCN (3 layers) ----------------------------------------------------
    const size_t NF = (size_t)NN * FEAT;
    fill_kernel<<<g1(NN, 256), dim3(256), 0, stream>>>(deg, 0.f, (size_t)NN);
    deg_kernel<<<g1(2 * (size_t)E, 256), dim3(256), 0, stream>>>(esrc, edst, deg, E);
    dinv_kernel<<<g1(NN, 256), dim3(256), 0, stream>>>(deg, NN);
    scale_copy_kernel<<<g1(NF, 256), dim3(256), 0, stream>>>(outA, xa, 0.25f, NF);

    float* cur = xa;
    float* nxt = xb;
    for (int l = 0; l < 3; ++l) {
        fill_kernel<<<g1(NF, 256), dim3(256), 0, stream>>>(nxt, 0.f, NF);
        scatter_kernel<<<g1(2 * (size_t)E * 32, 256), dim3(256), 0, stream>>>(
            esrc, edst, deg, cur, nxt, E);
        axpy_kernel<<<g1(NF, 256), dim3(256), 0, stream>>>(outA, nxt, 0.25f, NF);
        float* t = cur; cur = nxt; nxt = t;
    }

    // 4) pred = z_m @ O^T, O = outA[NUM_MASHUP:] ; [1024 x 50000] ---------------
    // 50000 = 781 full 64-col groups + one 16-col tail.
    const int NFULL = (NUM_API / 64) * 64;           // 49984
    gemm_wmma_kernel<true, 4><<<dim3(BATCH / 32, (781 + 3) / 4), dim3(32, 4), 0, stream>>>(
        zm, FEAT, outA + (size_t)NUM_MASHUP * FEAT, FEAT,
        pred, NUM_API, FEAT, 0, NFULL, 1.f, nullptr, 0, 0.f);
    gemm_wmma_kernel<true, 1><<<dim3(BATCH / 32, 1), dim3(32, 1), 0, stream>>>(
        zm, FEAT, outA + (size_t)NUM_MASHUP * FEAT, FEAT,
        pred, NUM_API, FEAT, NFULL, NUM_API, 1.f, nullptr, 0, 0.f);
}
